// PointNetSetAbstractionMsg_81587198754953
// MI455X (gfx1250) — matrix operand order repaired and checked by reading, not compiled, not run on hardware
//
#include <hip/hip_runtime.h>
#include <hip/hip_bf16.h>

typedef __attribute__((ext_vector_type(16))) _Float16 v16h;
typedef __attribute__((ext_vector_type(8)))  float    v8f;

#define NB   8
#define NPTS 4096
#define NS   512
#define BIGV 1e10f

struct LayerBN {
  const float* b; const float* beta; const float* gamma; const float* mean; const float* var;
};

// ------------------------------------------------------------------
// Farthest point sampling: 1 block per batch, sequential 512 steps.
// Coordinates register-resident (16 pts/thread), LDS argmax tree with
// first-index tie-break to match jnp.argmax.
// ------------------------------------------------------------------
__global__ __launch_bounds__(256) void fps_kernel(const float* __restrict__ xyz,
                                                  float* __restrict__ newxyz_ws,
                                                  float* __restrict__ out)
{
  const int b   = blockIdx.x;
  const int tid = threadIdx.x;
  const float* X = xyz + (size_t)b * 3 * NPTS;
  __shared__ float sv[256];
  __shared__ int   si[256];

  float px[16], py[16], pz[16], dist[16];
#pragma unroll
  for (int i = 0; i < 16; ++i) {
    int n = i * 256 + tid;
    px[i] = X[n]; py[i] = X[NPTS + n]; pz[i] = X[2 * NPTS + n];
    dist[i] = BIGV;
  }
  int far = 0;
  for (int t = 0; t < NS; ++t) {
    float cx = X[far], cy = X[NPTS + far], cz = X[2 * NPTS + far];
    if (tid == 0) {
      out[(b * 3 + 0) * NS + t] = cx;
      out[(b * 3 + 1) * NS + t] = cy;
      out[(b * 3 + 2) * NS + t] = cz;
      float* nw = newxyz_ws + ((size_t)b * NS + t) * 3;
      nw[0] = cx; nw[1] = cy; nw[2] = cz;
    }
    float bv = -1.f; int bi = 0;
#pragma unroll
    for (int i = 0; i < 16; ++i) {
      float dx = px[i] - cx, dy = py[i] - cy, dz = pz[i] - cz;
      float d = dx * dx + dy * dy + dz * dz;
      dist[i] = fminf(dist[i], d);
      if (dist[i] > bv) { bv = dist[i]; bi = i * 256 + tid; }
    }
    sv[tid] = bv; si[tid] = bi;
    __syncthreads();
    for (int st = 128; st > 0; st >>= 1) {
      if (tid < st) {
        float v2 = sv[tid + st]; int i2 = si[tid + st];
        if (v2 > sv[tid] || (v2 == sv[tid] && i2 < si[tid])) { sv[tid] = v2; si[tid] = i2; }
      }
      __syncthreads();
    }
    far = si[0];
    __syncthreads();
  }
}

// ------------------------------------------------------------------
// Ball query: exact K-nearest-within-radius, padded with global argmin
// (matches reference top_k-of-masked semantics). 1 block per (b,s),
// grid.y = branch. 32 masked distances per thread in registers;
// repeated min-extraction, only winner recomputes its local min.
// ------------------------------------------------------------------
struct BQArgs { float r2[3]; int K[3]; unsigned off[3]; };

__global__ __launch_bounds__(128) void ball_query_kernel(const float* __restrict__ xyz,
                                                         const float* __restrict__ newxyz,
                                                         unsigned short* __restrict__ idxws,
                                                         BQArgs a)
{
  const int br  = blockIdx.y;
  const int bs  = blockIdx.x;
  const int b   = bs >> 9;
  const int tid = threadIdx.x;
  const int K   = a.K[br];
  const float r2 = a.r2[br];
  unsigned short* out = idxws + a.off[br] + (size_t)bs * K;

  __shared__ float sv[128];
  __shared__ int   si[128];

  const float qx = newxyz[bs * 3 + 0], qy = newxyz[bs * 3 + 1], qz = newxyz[bs * 3 + 2];
  const float* X = xyz + (size_t)b * 3 * NPTS;

  float dm[32];
  float rmin = 3.4e38f; int rmi = 0;
#pragma unroll
  for (int i = 0; i < 32; ++i) {
    int n = i * 128 + tid;
    float dx = X[n] - qx, dy = X[NPTS + n] - qy, dz = X[2 * NPTS + n] - qz;
    float d = dx * dx + dy * dy + dz * dz;
    if (d < rmin) { rmin = d; rmi = n; }
    dm[i] = (d <= r2) ? d : BIGV;
  }
  // global nearest (first-index tie-break)
  sv[tid] = rmin; si[tid] = rmi;
  __syncthreads();
  for (int st = 64; st > 0; st >>= 1) {
    if (tid < st) {
      float v = sv[tid + st]; int j = si[tid + st];
      if (v < sv[tid] || (v == sv[tid] && j < si[tid])) { sv[tid] = v; si[tid] = j; }
    }
    __syncthreads();
  }
  const int nearest = si[0];
  __syncthreads();

  float cval = 3.4e38f; int cidx = 0;
#pragma unroll
  for (int i = 0; i < 32; ++i)
    if (dm[i] < cval) { cval = dm[i]; cidx = i * 128 + tid; }

  for (int k = 0; k < K; ++k) {
    sv[tid] = cval; si[tid] = cidx;
    __syncthreads();
    for (int st = 64; st > 0; st >>= 1) {
      if (tid < st) {
        float v = sv[tid + st]; int j = si[tid + st];
        if (v < sv[tid] || (v == sv[tid] && j < si[tid])) { sv[tid] = v; si[tid] = j; }
      }
      __syncthreads();
    }
    float mv = sv[0]; int mi = si[0];
    __syncthreads();
    if (mv >= BIGV) {                      // ball exhausted: pad with nearest
      if (tid == 0)
        for (int kk = k; kk < K; ++kk) out[kk] = (unsigned short)nearest;
      break;                               // uniform branch
    }
    if (tid == 0) out[k] = (unsigned short)mi;
    if (tid == (mi & 127)) {
      const int slot = mi >> 7;
#pragma unroll
      for (int i = 0; i < 32; ++i) if (i == slot) dm[i] = BIGV;
      cval = 3.4e38f; cidx = 0;
#pragma unroll
      for (int i = 0; i < 32; ++i)
        if (dm[i] < cval) { cval = dm[i]; cidx = i * 128 + tid; }
    }
  }
}

// ------------------------------------------------------------------
// Weight staging: f32 [O,C] -> f16 [O,Cpad] (Cpad multiple of 32, zero pad)
// ------------------------------------------------------------------
struct PrepArgs { const float* w[9]; int O[9]; int C[9]; int CP[9]; int off[9]; };

__global__ void prep_weights(PrepArgs a, _Float16* __restrict__ dst)
{
  const int l  = blockIdx.x;
  const int CP = a.CP[l], C = a.C[l], off = a.off[l];
  const int tot = a.O[l] * CP;
  const float* w = a.w[l];
  for (int e = threadIdx.x; e < tot; e += blockDim.x) {
    int o = e / CP, c = e % CP;
    float v = (c < C) ? w[o * C + c] : 0.f;
    dst[off + e] = (_Float16)v;
  }
}

// ------------------------------------------------------------------
// One MLP layer as tiled WMMA GEMM: D[K,O] = act[K,Cpad] * w16[O,Cpad]^T.
// A fragment packed from LDS in contiguous k order (k = 16h + s); B fragment is one
// aligned 32B load from staged f16 weights in the same contiguous k order.
// BN+ReLU applied per-lane (column constant), result either stored to the
// next LDS activation buffer or max-pooled via uint atomicMax (ReLU >= 0).
// ------------------------------------------------------------------
__device__ __forceinline__ void mlp_layer(const _Float16* act, _Float16* dst, unsigned* pool,
                                          const _Float16* __restrict__ w16, LayerBN bn,
                                          int Krows, int Cpad, int O, int CpadOut, int nthreads)
{
  const int lane = threadIdx.x & 31;
  const int wave = threadIdx.x >> 5;
  const int nwv  = nthreads >> 5;
  const int MT = Krows >> 4, NT = O >> 4;
  const int n16 = lane & 15;
  const int hi  = (lane >= 16) ? 8 : 0;

  for (int job = wave; job < MT * NT; job += nwv) {
    const int tm = job / NT, tn = job % NT;
    const int arow = tm * 16 + n16;       // A: m = lane&15
    const int ocol = tn * 16 + n16;       // B/D column = lane&15
    v8f acc = {};
    for (int kb = 0; kb < Cpad; kb += 32) {
      union { v16h h; unsigned u[8]; } A;
      const unsigned* ap = (const unsigned*)(act + (size_t)arow * Cpad + kb);
#pragma unroll
      for (int v = 0; v < 8; ++v)         // K-halves: 2v + 2*hi (contiguous)
        A.u[v] = ap[v + hi];
      const v16h Bf = *(const v16h*)(w16 + (size_t)ocol * Cpad + kb + ((lane >= 16) ? 16 : 0));
      acc = __builtin_amdgcn_wmma_f32_16x16x32_f16(false, A.h, false, Bf,
                                                   (short)0, acc, false, false);
    }
    const float inv = rsqrtf(bn.var[ocol] + 1e-5f);
    const float sc  = bn.gamma[ocol] * inv;
    const float sh  = (bn.b[ocol] - bn.mean[ocol]) * sc + bn.beta[ocol];
    if (dst) {
#pragma unroll
      for (int i = 0; i < 8; ++i) {       // D: row = tm*16 + i + hi
        float y = fmaxf(acc[i] * sc + sh, 0.f);
        dst[(size_t)(tm * 16 + hi + i) * CpadOut + ocol] = (_Float16)y;
      }
    } else {
      float m = 0.f;
#pragma unroll
      for (int i = 0; i < 8; ++i) m = fmaxf(m, fmaxf(acc[i] * sc + sh, 0.f));
      atomicMax(&pool[ocol], __float_as_uint(m));   // ds_max_u32; valid: relu >= 0
    }
  }
}

// ------------------------------------------------------------------
// Per-(b,s) group kernel: gather K neighbors' features to LDS (f16),
// run 3 WMMA MLP layers ping-ponging LDS buffers, max-pool over K.
// ------------------------------------------------------------------
template <int K, int O0, int O1, int O2, int CP0, int CP1, int CP2, int THREADS>
__global__ __launch_bounds__(THREADS) void group_kernel(
    const float* __restrict__ xyz, const float* __restrict__ points,
    const float* __restrict__ newxyz, const unsigned short* __restrict__ gidx,
    const _Float16* __restrict__ w0, const _Float16* __restrict__ w1,
    const _Float16* __restrict__ w2,
    LayerBN bn0, LayerBN bn1, LayerBN bn2,
    float* __restrict__ out, int chbase)
{
  constexpr int BUFA = K * ((CP0 > CP2) ? CP0 : CP2);
  __shared__ __align__(16) _Float16 bufA[BUFA];
  __shared__ __align__(16) _Float16 bufB[K * CP1];
  __shared__ unsigned pool[O2];
  __shared__ unsigned short sid[K];

  const int tid = threadIdx.x;
  const int bs  = blockIdx.x;
  const int b   = bs >> 9, s = bs & 511;

  for (int k = tid; k < K; k += THREADS) sid[k] = gidx[(size_t)bs * K + k];
  for (int o = tid; o < O2; o += THREADS) pool[o] = 0u;
  __builtin_prefetch(w0, 0, 1);
  __syncthreads();

  const float nx = newxyz[bs * 3 + 0], ny = newxyz[bs * 3 + 1], nz = newxyz[bs * 3 + 2];
  const float* P = points + (size_t)b * 64 * NPTS;
  const float* X = xyz + (size_t)b * 3 * NPTS;

  for (int e = tid; e < K * CP0; e += THREADS) {
    int k = e & (K - 1);
    int c = e / K;
    int id = sid[k];
    float v;
    if      (c < 64)  v = P[(size_t)c * NPTS + id];
    else if (c == 64) v = X[id] - nx;
    else if (c == 65) v = X[NPTS + id] - ny;
    else if (c == 66) v = X[2 * NPTS + id] - nz;
    else              v = 0.f;
    bufA[k * CP0 + c] = (_Float16)v;
  }
  __syncthreads();
  mlp_layer(bufA, bufB, nullptr, w0, bn0, K, CP0, O0, CP1, THREADS);
  __syncthreads();
  mlp_layer(bufB, bufA, nullptr, w1, bn1, K, CP1, O1, CP2, THREADS);
  __syncthreads();
  mlp_layer(bufA, nullptr, pool, w2, bn2, K, CP2, O2, 0, THREADS);
  __syncthreads();

  float* OUT = out + NB * 3 * NS;   // second tuple element [B,320,S]
  for (int o = tid; o < O2; o += THREADS)
    OUT[((size_t)b * 320 + chbase + o) * NS + s] = __uint_as_float(pool[o]);
}

// ------------------------------------------------------------------
extern "C" void kernel_launch(void* const* d_in, const int* in_sizes, int n_in,
                              void* d_out, int out_size, void* d_ws, size_t ws_size,
                              hipStream_t stream)
{
  (void)in_sizes; (void)n_in; (void)out_size; (void)ws_size;
  const float* xyz    = (const float*)d_in[0];
  const float* points = (const float*)d_in[1];
  // params pytree: dict keys sorted: b, beta, gamma, mean, var, w
  auto din = [&](int br, int ly, int which) -> const float* {
    return (const float*)d_in[2 + (br * 3 + ly) * 6 + which];
  };
  auto bn = [&](int br, int ly) -> LayerBN {
    return LayerBN{ din(br, ly, 0), din(br, ly, 1), din(br, ly, 2), din(br, ly, 3), din(br, ly, 4) };
  };

  // workspace layout
  char* ws = (char*)d_ws;
  _Float16*       w16   = (_Float16*)ws;                 // 49152 halves = 98304 B
  float*          nxyz  = (float*)(ws + 98304);          // [B,S,3] = 49152 B
  unsigned short* idxws = (unsigned short*)(ws + 147456);

  float* out = (float*)d_out;

  static const int Os[9]  = {32, 32, 64,  64, 64, 128,  64, 96, 128};
  static const int Cs[9]  = {67, 32, 32,  67, 64, 64,   67, 64, 96};
  static const int CPs[9] = {96, 32, 32,  96, 64, 64,   96, 64, 96};

  PrepArgs pa;
  int woff[9];
  {
    int off = 0;
    for (int l = 0; l < 9; ++l) {
      pa.w[l] = din(l / 3, l % 3, 5);
      pa.O[l] = Os[l]; pa.C[l] = Cs[l]; pa.CP[l] = CPs[l]; pa.off[l] = off;
      woff[l] = off;
      off += Os[l] * CPs[l];
    }
  }
  prep_weights<<<9, 128, 0, stream>>>(pa, w16);

  fps_kernel<<<NB, 256, 0, stream>>>(xyz, nxyz, out);

  BQArgs ba;
  ba.r2[0] = 0.01f; ba.r2[1] = 0.04f; ba.r2[2] = 0.16f;
  ba.K[0] = 16; ba.K[1] = 32; ba.K[2] = 128;
  ba.off[0] = 0;
  ba.off[1] = NB * NS * 16;
  ba.off[2] = NB * NS * 16 + NB * NS * 32;
  ball_query_kernel<<<dim3(NB * NS, 3), 128, 0, stream>>>(xyz, nxyz, idxws, ba);

  group_kernel<16, 32, 32, 64, 96, 32, 32, 64><<<NB * NS, 64, 0, stream>>>(
      xyz, points, nxyz, idxws + ba.off[0],
      w16 + woff[0], w16 + woff[1], w16 + woff[2],
      bn(0, 0), bn(0, 1), bn(0, 2), out, 0);

  group_kernel<32, 64, 64, 128, 96, 64, 64, 128><<<NB * NS, 128, 0, stream>>>(
      xyz, points, nxyz, idxws + ba.off[1],
      w16 + woff[3], w16 + woff[4], w16 + woff[5],
      bn(1, 0), bn(1, 1), bn(1, 2), out, 64);

  group_kernel<128, 64, 96, 128, 96, 64, 96, 256><<<NB * NS, 256, 0, stream>>>(
      xyz, points, nxyz, idxws + ba.off[2],
      w16 + woff[6], w16 + woff[7], w16 + woff[8],
      bn(2, 0), bn(2, 1), bn(2, 2), out, 192);
}
